// HSTULayer_81217831567691
// MI455X (gfx1250) — matrix operand order repaired and checked by reading, not compiled, not run on hardware
//
#include <hip/hip_runtime.h>
#include <hip/hip_bf16.h>

typedef __attribute__((ext_vector_type(16))) __bf16 v16bf;
typedef __attribute__((ext_vector_type(8)))  __bf16 v8bf;
typedef __attribute__((ext_vector_type(8)))  float  v8f;
typedef __attribute__((ext_vector_type(4)))  int    v4i;

#define D_MODEL  512
#define NHEADS   8
#define DQK      64
#define DVDIM    64
#define PROJ_OUT 2048
#define FFN_HID  2048
#define SEQL     2048
#define BATCH    4
#define NROWS    (BATCH * SEQL)   // 8192
#define ATT_SCALE 0.125f
#define LN_EPS   1e-5f

// ---- CDNA5 async global->LDS path (probe-guarded; sync fallback) ----------
#if defined(__has_builtin)
#if __has_builtin(__builtin_amdgcn_global_load_async_to_lds_b128)
#define HAS_ASYNC_LDS 1
#endif
#if __has_builtin(__builtin_amdgcn_s_wait_asynccnt)
#define HAS_WAIT_ASYNC 1
#endif
#endif

typedef __attribute__((address_space(1))) v4i  as1_v4i;
typedef __attribute__((address_space(3))) v4i  as3_v4i;
typedef __attribute__((address_space(1))) void as1_void;
typedef __attribute__((address_space(3))) void as3_void;

__device__ __forceinline__ v16bf cat8(v8bf lo, v8bf hi) {
    return __builtin_shufflevector(lo, hi, 0,1,2,3,4,5,6,7,8,9,10,11,12,13,14,15);
}
__device__ __forceinline__ v8bf ld8(const __bf16* p) { return *(const v8bf*)p; }

// 16-byte global -> LDS copy; async on CDNA5, VGPR round-trip otherwise.
__device__ __forceinline__ void cp16_g2s(const __bf16* g, __bf16* l) {
#if defined(HAS_ASYNC_LDS)
    __builtin_amdgcn_global_load_async_to_lds_b128(
        (as1_v4i*)(as1_void*)(void*)g,
        (as3_v4i*)(as3_void*)(void*)l, 0, 0);
#else
    *(v8bf*)l = ld8(g);
#endif
}

template <int N>
__device__ __forceinline__ void wait_asynccnt() {
#if defined(HAS_ASYNC_LDS)
#if defined(HAS_WAIT_ASYNC)
    __builtin_amdgcn_s_wait_asynccnt(N);
#else
    asm volatile("s_wait_asynccnt %0" ::"n"(N) : "memory");
#endif
#endif
}

// ---------------------------------------------------------------------------
// Weight transpose + f32 -> bf16 convert:  Wt[n*K + k] = (bf16) W[k*N + n]
// ---------------------------------------------------------------------------
__global__ __launch_bounds__(256)
void transpose_to_bf16(const float* __restrict__ W, __bf16* __restrict__ Wt,
                       int K, int N) {
    __shared__ float t[32][33];
    const int nb = blockIdx.x * 32, kb = blockIdx.y * 32;
    const int tx = threadIdx.x & 31, ty = threadIdx.x >> 5;   // ty: 0..7
#pragma unroll
    for (int i = 0; i < 4; ++i)
        t[ty * 4 + i][tx] = W[(size_t)(kb + ty * 4 + i) * N + nb + tx];
    __syncthreads();
#pragma unroll
    for (int i = 0; i < 4; ++i)
        Wt[(size_t)(nb + ty * 4 + i) * K + kb + tx] = (__bf16)t[tx][ty * 4 + i];
}

// ---------------------------------------------------------------------------
// Row LayerNorm (D=512) -> bf16 output. One block (256 thr) per row.
// ---------------------------------------------------------------------------
__global__ __launch_bounds__(256)
void layernorm_bf16(const float* __restrict__ x, const float* __restrict__ g,
                    const float* __restrict__ be, __bf16* __restrict__ out) {
    __shared__ float red[256];
    const int row = blockIdx.x;
    const int tid = threadIdx.x;
    const float* xr = x + (size_t)row * D_MODEL;
    const float x0 = xr[tid], x1 = xr[tid + 256];
    red[tid] = x0 + x1;
    __syncthreads();
    for (int o = 128; o > 0; o >>= 1) {
        if (tid < o) red[tid] += red[tid + o];
        __syncthreads();
    }
    const float mean = red[0] * (1.f / D_MODEL);
    __syncthreads();
    const float d0 = x0 - mean, d1 = x1 - mean;
    red[tid] = d0 * d0 + d1 * d1;
    __syncthreads();
    for (int o = 128; o > 0; o >>= 1) {
        if (tid < o) red[tid] += red[tid + o];
        __syncthreads();
    }
    const float var = red[0] * (1.f / D_MODEL);
    const float rstd = rsqrtf(var + LN_EPS);
    __bf16* orow = out + (size_t)row * D_MODEL;
    orow[tid]       = (__bf16)(d0 * rstd * g[tid]       + be[tid]);
    orow[tid + 256] = (__bf16)(d1 * rstd * g[tid + 256] + be[tid + 256]);
}

// ---------------------------------------------------------------------------
// WMMA GEMM, double-buffered async LDS staging.
// C[M,N] = A[M,K](bf16) @ Bt[N,K](bf16)^T (+bias, relu, residual).
// Block tile 128x128, BK=32, 256 threads = 8 waves (4x2), wave = 32x64.
// ---------------------------------------------------------------------------
template <bool RELU, bool RESID, bool OUTBF>
__global__ __launch_bounds__(256)
void gemm_wmma(const __bf16* __restrict__ A, const __bf16* __restrict__ Bt,
               const float* __restrict__ bias, const float* __restrict__ resid,
               float* __restrict__ outF, __bf16* __restrict__ outB,
               int M, int N, int K) {
    __shared__ __align__(16) __bf16 sA[2][128 * 32];
    __shared__ __align__(16) __bf16 sB[2][128 * 32];
    const int tid  = threadIdx.x;
    const int lane = tid & 31;
    const int wave = tid >> 5;
    const int wm = wave >> 1;   // 0..3 -> 32-row band
    const int wn = wave & 1;    // 0..1 -> 64-col band
    const int blockM = blockIdx.y * 128;
    const int blockN = blockIdx.x * 128;
    const int l15 = lane & 15;
    const int kb  = (lane < 16) ? 0 : 8;
    const int kb2 = (lane < 16) ? 0 : 16;
    const int hi8 = (lane >> 4) << 3;

    const v8f zero = {0.f, 0.f, 0.f, 0.f, 0.f, 0.f, 0.f, 0.f};
    v8f acc[2][4];
#pragma unroll
    for (int i = 0; i < 2; ++i)
#pragma unroll
        for (int j = 0; j < 4; ++j) acc[i][j] = zero;

    const int kTiles = K >> 5;
    const int lrow = tid >> 2;             // 0..63
    const int lcol = (tid & 3) << 3;       // 0,8,16,24

    auto stage = [&](int kt, int buf) {
        const int k0 = kt << 5;
#pragma unroll
        for (int p = 0; p < 2; ++p) {
            const int r = lrow + p * 64;
            cp16_g2s(&A[(size_t)(blockM + r) * K + k0 + lcol],
                     &sA[buf][r * 32 + lcol]);
            cp16_g2s(&Bt[(size_t)(blockN + r) * K + k0 + lcol],
                     &sB[buf][r * 32 + lcol]);
        }
    };

    stage(0, 0);   // prologue: 4 async (or sync) 16B copies per thread

    for (int kt = 0; kt < kTiles; ++kt) {
        const int cur = kt & 1;
        if (kt + 1 < kTiles) {
            stage(kt + 1, cur ^ 1);   // prefetch next tile into other buffer
            wait_asynccnt<4>();       // retire current tile's 4 copies
        } else {
            wait_asynccnt<0>();
        }
        __syncthreads();

        v16bf af[2], bfr[4];
#pragma unroll
        for (int fm = 0; fm < 2; ++fm) {
            const int r = wm * 32 + fm * 16 + l15;
            af[fm] = cat8(ld8(&sA[cur][r * 32 + kb2]),
                          ld8(&sA[cur][r * 32 + kb2 + 8]));
        }
#pragma unroll
        for (int fn = 0; fn < 4; ++fn) {
            const int c = wn * 64 + fn * 16 + l15;
            bfr[fn] = cat8(ld8(&sB[cur][c * 32 + kb2]),
                           ld8(&sB[cur][c * 32 + kb2 + 8]));
        }
#pragma unroll
        for (int fm = 0; fm < 2; ++fm)
#pragma unroll
            for (int fn = 0; fn < 4; ++fn)
                acc[fm][fn] = __builtin_amdgcn_wmma_f32_16x16x32_bf16(
                    false, af[fm], false, bfr[fn], (short)0, acc[fm][fn],
                    false, false);
        __syncthreads();
    }

#pragma unroll
    for (int fm = 0; fm < 2; ++fm) {
#pragma unroll
        for (int fn = 0; fn < 4; ++fn) {
            const int n = blockN + wn * 64 + fn * 16 + l15;
            const float bv = bias[n];
#pragma unroll
            for (int r = 0; r < 8; ++r) {
                const int m = blockM + wm * 32 + fm * 16 + hi8 + r;
                float v = acc[fm][fn][r] + bv;
                if (RELU) v = v > 0.f ? v : 0.f;
                if (RESID) v += resid[(size_t)m * N + n];
                if (OUTBF) outB[(size_t)m * N + n] = (__bf16)v;
                else       outF[(size_t)m * N + n] = v;
            }
        }
    }
}

// ---------------------------------------------------------------------------
// Flash attention with causal mask + rel-pos bias + sigmoid(u) gating.
// Grid (L/64, H, B); 128 threads = 4 waves, each wave owns 16 q-rows.
// p layout: [B*L, 2048] bf16 where cols: q | k | u | v (each H*64).
// ---------------------------------------------------------------------------
__global__ __launch_bounds__(128)
void attention_kernel(const __bf16* __restrict__ p, const float* __restrict__ rel,
                      __bf16* __restrict__ obuf) {
    __shared__ __align__(16) __bf16 sK[32 * 64];       // [kpos][dqk]
    __shared__ __align__(16) __bf16 sVt[64 * 32];      // [dv][kpos]
    __shared__ __align__(16) __bf16 sP[4 * 16 * 32];   // per-wave P tile
    const int tid  = threadIdx.x;
    const int lane = tid & 31;
    const int wave = tid >> 5;
    const int qbase = blockIdx.x * 64;
    const int h = blockIdx.y;
    const int b = blockIdx.z;
    const size_t rowbase = (size_t)b * SEQL;
    const size_t hLL = (size_t)h * SEQL * SEQL;

    const int l15 = lane & 15;
    const int kb  = (lane < 16) ? 0 : 8;
    const int kb2 = (lane < 16) ? 0 : 16;
    const int hi8 = (lane >> 4) << 3;

    // Q fragments for this wave's 16 rows (16x64 -> two 16x32 A-frags)
    const int qrowA = qbase + wave * 16 + l15;
    const __bf16* qp = p + (rowbase + qrowA) * PROJ_OUT + h * DQK;
    v16bf aq[2];
    aq[0] = cat8(ld8(qp + kb2),      ld8(qp + kb2 + 8));
    aq[1] = cat8(ld8(qp + 32 + kb2), ld8(qp + 32 + kb2 + 8));

    const v8f zero = {0.f, 0.f, 0.f, 0.f, 0.f, 0.f, 0.f, 0.f};
    float mrow[8], lrow[8];
    v8f oacc[4];
#pragma unroll
    for (int r = 0; r < 8; ++r) { mrow[r] = -1e30f; lrow[r] = 0.f; }
#pragma unroll
    for (int fv = 0; fv < 4; ++fv) oacc[fv] = zero;

    const int nkt = (qbase >> 5) + 2;      // causal: k-tiles up to qbase+63
    const int srow = tid >> 2;             // 0..31
    const int scol = (tid & 3) << 4;       // 0,16,32,48

    for (int kt = 0; kt < nkt; ++kt) {
        const int kbase = kt << 5;
        {   // stage K tile [32][64] (async on CDNA5) and V^T tile [64][32]
            const size_t krow = (rowbase + kbase + srow) * PROJ_OUT;
            const __bf16* kp = p + krow + D_MODEL + h * DQK + scol;
            cp16_g2s(kp,     &sK[srow * 64 + scol]);
            cp16_g2s(kp + 8, &sK[srow * 64 + scol + 8]);
            const __bf16* vp = p + krow + 3 * D_MODEL + h * DVDIM + scol;
            const v8bf v0 = ld8(vp), v1 = ld8(vp + 8);
#pragma unroll
            for (int j = 0; j < 8; ++j) {
                sVt[(scol + j) * 32 + srow]     = v0[j];
                sVt[(scol + 8 + j) * 32 + srow] = v1[j];
            }
        }
        // prefetch next tile's rel-pos-bias rows (dominant HBM stream)
        if (kt + 1 < nkt) {
            const float* nb = rel + hLL +
                (size_t)(qbase + wave * 16 + l15) * SEQL + (kbase + 32) +
                ((lane >> 4) << 4);
            __builtin_prefetch(nb, 0, 1);
        }
        wait_asynccnt<0>();
        __syncthreads();

        // S = Q @ K^T  (16x32), two 16x16 fragments over k-positions
        float sv[2][8];
#pragma unroll
        for (int fn = 0; fn < 2; ++fn) {
            v8f s = zero;
#pragma unroll
            for (int ks = 0; ks < 2; ++ks) {
                const int c = fn * 16 + l15;   // k-position column
                v16bf bk = cat8(ld8(&sK[c * 64 + ks * 32 + kb2]),
                                ld8(&sK[c * 64 + ks * 32 + kb2 + 8]));
                s = __builtin_amdgcn_wmma_f32_16x16x32_bf16(
                    false, aq[ks], false, bk, (short)0, s, false, false);
            }
            const int kg = kbase + fn * 16 + l15;
#pragma unroll
            for (int r = 0; r < 8; ++r) {
                const int qg = qbase + wave * 16 + hi8 + r;
                const float bias = rel[hLL + (size_t)qg * SEQL + kg];
                const float val = s[r] * ATT_SCALE + bias;
                sv[fn][r] = (kg <= qg) ? val : -1e30f;
            }
        }

        // online softmax (row reductions across the 16-lane half via shfl_xor)
        float alpha[8];
#pragma unroll
        for (int r = 0; r < 8; ++r) {
            float rm = fmaxf(sv[0][r], sv[1][r]);
            rm = fmaxf(rm, __shfl_xor(rm, 1, 32));
            rm = fmaxf(rm, __shfl_xor(rm, 2, 32));
            rm = fmaxf(rm, __shfl_xor(rm, 4, 32));
            rm = fmaxf(rm, __shfl_xor(rm, 8, 32));
            const float mnew = fmaxf(mrow[r], rm);
            const float a  = __expf(mrow[r] - mnew);
            const float p0 = __expf(sv[0][r] - mnew);
            const float p1 = __expf(sv[1][r] - mnew);
            float rs = p0 + p1;
            rs += __shfl_xor(rs, 1, 32);
            rs += __shfl_xor(rs, 2, 32);
            rs += __shfl_xor(rs, 4, 32);
            rs += __shfl_xor(rs, 8, 32);
            lrow[r] = lrow[r] * a + rs;
            mrow[r] = mnew;
            alpha[r] = a;
            const int prow = hi8 + r;   // relayout P into A-frag order
            sP[wave * 512 + prow * 32 + l15]      = (__bf16)p0;
            sP[wave * 512 + prow * 32 + 16 + l15] = (__bf16)p1;
        }
#pragma unroll
        for (int fv = 0; fv < 4; ++fv)
#pragma unroll
            for (int r = 0; r < 8; ++r) oacc[fv][r] *= alpha[r];

        // O += P(16x32) @ V(32x64)
        const __bf16* pw = &sP[wave * 512 + l15 * 32];
        const v16bf pa = cat8(*(const v8bf*)(pw + kb2),
                              *(const v8bf*)(pw + kb2 + 8));
#pragma unroll
        for (int fv = 0; fv < 4; ++fv) {
            const int dv = fv * 16 + l15;
            v16bf bv = cat8(ld8(&sVt[dv * 32 + kb2]),
                            ld8(&sVt[dv * 32 + kb2 + 8]));
            oacc[fv] = __builtin_amdgcn_wmma_f32_16x16x32_bf16(
                false, pa, false, bv, (short)0, oacc[fv], false, false);
        }
        __syncthreads();
    }

    // epilogue: normalize, gate by sigmoid(u), emit bf16 [B*L, 512]
#pragma unroll
    for (int fv = 0; fv < 4; ++fv) {
        const int d = fv * 16 + l15;
#pragma unroll
        for (int r = 0; r < 8; ++r) {
            const int qg = qbase + wave * 16 + hi8 + r;
            const size_t row = rowbase + qg;
            const float u =
                (float)p[row * PROJ_OUT + 2 * D_MODEL + h * DVDIM + d];
            const float o = oacc[fv][r] / lrow[r];
            const float gt = o * (1.f / (1.f + __expf(-u)));
            obuf[row * D_MODEL + h * DVDIM + d] = (__bf16)gt;
        }
    }
}

// ---------------------------------------------------------------------------
extern "C" void kernel_launch(void* const* d_in, const int* in_sizes, int n_in,
                              void* d_out, int out_size, void* d_ws,
                              size_t ws_size, hipStream_t stream) {
    (void)in_sizes; (void)n_in; (void)out_size; (void)ws_size;
    const float* x   = (const float*)d_in[0];
    const float* rel = (const float*)d_in[1];
    const float* W1  = (const float*)d_in[2];
    const float* b1  = (const float*)d_in[3];
    const float* W2  = (const float*)d_in[4];
    const float* b2  = (const float*)d_in[5];
    const float* Wf1 = (const float*)d_in[6];
    const float* bf1 = (const float*)d_in[7];
    const float* Wf2 = (const float*)d_in[8];
    const float* bf2 = (const float*)d_in[9];
    const float* g1  = (const float*)d_in[10];
    const float* be1 = (const float*)d_in[11];
    const float* g2  = (const float*)d_in[12];
    const float* be2 = (const float*)d_in[13];
    float* out = (float*)d_out;

    char* ws = (char*)d_ws;
    size_t off = 0;
    __bf16* hbf  = (__bf16*)(ws + off); off += (size_t)NROWS * D_MODEL * 2;
    __bf16* pbuf = (__bf16*)(ws + off); off += (size_t)NROWS * PROJ_OUT * 2;
    __bf16* obuf = (__bf16*)(ws + off); off += (size_t)NROWS * D_MODEL * 2;
    float*  y1   = (float*) (ws + off); off += (size_t)NROWS * D_MODEL * 4;
    __bf16* h2   = (__bf16*)(ws + off); off += (size_t)NROWS * D_MODEL * 2;
    __bf16* ffh  = (__bf16*)(ws + off); off += (size_t)NROWS * FFN_HID * 2;
    __bf16* w1t  = (__bf16*)(ws + off); off += (size_t)PROJ_OUT * D_MODEL * 2;
    __bf16* w2t  = (__bf16*)(ws + off); off += (size_t)D_MODEL * D_MODEL * 2;
    __bf16* wf1t = (__bf16*)(ws + off); off += (size_t)FFN_HID * D_MODEL * 2;
    __bf16* wf2t = (__bf16*)(ws + off); off += (size_t)D_MODEL * FFN_HID * 2;

    // one-time weight transpose + bf16 convert
    transpose_to_bf16<<<dim3(PROJ_OUT / 32, D_MODEL / 32), 256, 0, stream>>>(
        W1, w1t, D_MODEL, PROJ_OUT);
    transpose_to_bf16<<<dim3(D_MODEL / 32, D_MODEL / 32), 256, 0, stream>>>(
        W2, w2t, D_MODEL, D_MODEL);
    transpose_to_bf16<<<dim3(FFN_HID / 32, D_MODEL / 32), 256, 0, stream>>>(
        Wf1, wf1t, D_MODEL, FFN_HID);
    transpose_to_bf16<<<dim3(D_MODEL / 32, FFN_HID / 32), 256, 0, stream>>>(
        Wf2, wf2t, FFN_HID, D_MODEL);

    // LN1 -> q,k,u,v projection
    layernorm_bf16<<<NROWS, 256, 0, stream>>>(x, g1, be1, hbf);
    gemm_wmma<false, false, true>
        <<<dim3(PROJ_OUT / 128, NROWS / 128), 256, 0, stream>>>(
            hbf, w1t, b1, nullptr, nullptr, pbuf, NROWS, PROJ_OUT, D_MODEL);

    // attention + gating
    attention_kernel<<<dim3(SEQL / 64, NHEADS, BATCH), 128, 0, stream>>>(
        pbuf, rel, obuf);

    // out-proj + residual
    gemm_wmma<false, true, false>
        <<<dim3(D_MODEL / 128, NROWS / 128), 256, 0, stream>>>(
            obuf, w2t, b2, x, y1, nullptr, NROWS, D_MODEL, D_MODEL);

    // LN2 -> FFN
    layernorm_bf16<<<NROWS, 256, 0, stream>>>(y1, g2, be2, h2);
    gemm_wmma<true, false, true>
        <<<dim3(FFN_HID / 128, NROWS / 128), 256, 0, stream>>>(
            h2, wf1t, bf1, nullptr, nullptr, ffh, NROWS, FFN_HID, D_MODEL);
    gemm_wmma<false, true, false>
        <<<dim3(D_MODEL / 128, NROWS / 128), 256, 0, stream>>>(
            ffh, wf2t, bf2, y1, out, nullptr, NROWS, D_MODEL, FFN_HID);
}
